// VibrationGAT_29746943492467
// MI455X (gfx1250) — compile-verified, matrix operand order checked by reading, not run on hardware
//
#include <hip/hip_runtime.h>
#include <hip/hip_bf16.h>
#include <math.h>

// ---- problem constants (match reference) ----
#define N_NODES   20000
#define N_EDGES   200000
#define N_GRAPHS  256
#define N_FEAT    24
#define HIDDEN    128
#define HEADS     8
#define N_CLASSES 4
#define NEG_SLOPE 0.2f
#define EPS_SM    1e-16f
#define D1        (HIDDEN * HEADS)   // 1024
#define E_TOT     (N_EDGES + N_NODES)

#define USE_ASYNC 1   // gfx1250 async global->LDS staging (set 0 to fall back)

typedef __attribute__((ext_vector_type(16))) __bf16 bf16x16;
typedef __attribute__((ext_vector_type(8)))  float  floatx8;

__device__ __forceinline__ unsigned short f2bf(float f) {
  unsigned int u = __float_as_uint(f);
  u += 0x7FFFu + ((u >> 16) & 1u);   // round to nearest even
  return (unsigned short)(u >> 16);
}

// 16-byte copy global -> LDS. Async path uses the CDNA5 ASYNCcnt-tracked
// DMA (GLOBAL_LOAD_ASYNC_TO_LDS_B128: VDST = LDS byte offset, VADDR = addr).
__device__ __forceinline__ void copy16B_to_lds(unsigned short* lds,
                                               const unsigned short* g) {
#if USE_ASYNC
  unsigned off = (unsigned)(size_t)lds;   // low 32 bits of flat addr = LDS offset
  asm volatile("global_load_async_to_lds_b128 %0, %1, off"
               :: "v"(off), "v"(g) : "memory");
#else
  *(int4*)lds = *(const int4*)g;
#endif
}

__device__ __forceinline__ void async_wait() {
#if USE_ASYNC
  asm volatile("s_wait_asynccnt 0x0" ::: "memory");
#endif
}

// =====================================================================
// Pre-pack kernels: fp32 row-major -> bf16 WMMA fragment-major tiles.
// Tile = 16(rows/cols) x 32(K) bf16 = 512 ushort = 1KB; per-lane 32B
// contiguous: P[tile][lane][elem], k = ((e<8)?e:e+8) + (lane>>4)*8.
// A tiles ordered [rowTile][kTile]; B tiles ordered [kTile][colTile].
// =====================================================================
__global__ void k_pack_A(const float* __restrict__ X, unsigned short* __restrict__ P,
                         int M, int K, int MT, int KT) {
  long long i = (long long)blockIdx.x * blockDim.x + threadIdx.x;
  long long total = (long long)MT * KT * 512;
  if (i >= total) return;
  int e    = (int)(i & 15);
  int lane = (int)((i >> 4) & 31);
  long long t = i >> 9;
  int st = (int)(t % KT);
  int rt = (int)(t / KT);
  int row = rt * 16 + (lane & 15);
  int k   = st * 32 + ((e < 8) ? e : e + 8) + (lane >> 4) * 8;
  float v = (row < M && k < K) ? X[(size_t)row * K + k] : 0.f;
  P[i] = f2bf(v);
}

__global__ void k_pack_B(const float* __restrict__ X, unsigned short* __restrict__ P,
                         int K, int N, int KT, int NT) {
  long long i = (long long)blockIdx.x * blockDim.x + threadIdx.x;
  long long total = (long long)KT * NT * 512;
  if (i >= total) return;
  int e    = (int)(i & 15);
  int lane = (int)((i >> 4) & 31);
  long long t = i >> 9;
  int nt = (int)(t % NT);
  int st = (int)(t / NT);
  int col = nt * 16 + (lane & 15);
  int k   = st * 32 + ((e < 8) ? e : e + 8) + (lane >> 4) * 8;
  float v = (k < K && col < N) ? X[(size_t)k * N + col] : 0.f;
  P[i] = f2bf(v);
}

// =====================================================================
// GEMM on pre-packed bf16 fragments: C[M,N] = A @ B, fp32 accumulate/out.
// Block = 256 threads = 8 waves (4m x 2n); wave owns 32x64 output
// (2x4 WMMA tiles -> 8 v_wmma per K-step). Staging: each wave async-DMAs
// one A tile + one B tile (1KB each) into LDS per K-step.
// =====================================================================
static constexpr int MT_BLK = 8;   // 8 row tiles  = 128 rows per block
static constexpr int NT_BLK = 8;   // 8 col tiles  = 128 cols per block

__global__ __launch_bounds__(256)
void k_gemm_packed(const unsigned short* __restrict__ Apk,  // [MT][KT][512]
                   const unsigned short* __restrict__ Bpk,  // [KT][NT][512]
                   float* __restrict__ C,
                   int MT, int KT, int NT, int M, int N) {
  __shared__ __align__(32) unsigned short Als[MT_BLK][512];  // 8 KB
  __shared__ __align__(32) unsigned short Bls[NT_BLK][512];  // 8 KB

  const int bm   = blockIdx.y;
  const int bn   = blockIdx.x;
  const int tid  = threadIdx.x;
  const int wid  = tid >> 5;
  const int lane = tid & 31;
  const int wm   = wid & 3;       // wave m (0..3) -> m tiles 2*wm, 2*wm+1
  const int wn   = wid >> 2;      // wave n (0..1) -> n tiles 4*wn .. 4*wn+3
  const int half = lane >> 4;
  const int l16  = lane & 15;

  // staging assignment: wave `wid` copies A tile `wid` and B tile `wid`;
  // lane copies its own 32B chunk (2 x 16B).
  const int rt_cp = bm * MT_BLK + wid;
  const int nt_cp = bn * NT_BLK + wid;

  floatx8 acc[2][4] = {};

  for (int kt = 0; kt < KT; ++kt) {
    if (kt) __syncthreads();            // previous compute done before overwrite
    if (rt_cp < MT) {                   // wave-uniform guard
      const unsigned short* ga = Apk + (((size_t)rt_cp * KT + kt) << 9) + lane * 16;
      copy16B_to_lds(&Als[wid][lane * 16], ga);
      copy16B_to_lds(&Als[wid][lane * 16 + 8], ga + 8);
    }
    if (nt_cp < NT) {
      const unsigned short* gb = Bpk + (((size_t)kt * NT + nt_cp) << 9) + lane * 16;
      copy16B_to_lds(&Bls[wid][lane * 16], gb);
      copy16B_to_lds(&Bls[wid][lane * 16 + 8], gb + 8);
    }
    async_wait();                       // own ASYNCcnt drained
    __syncthreads();                    // all waves' copies visible

    bf16x16 a0 = *(const bf16x16*)&Als[wm * 2 + 0][lane * 16];
    bf16x16 a1 = *(const bf16x16*)&Als[wm * 2 + 1][lane * 16];
    bf16x16 b0 = *(const bf16x16*)&Bls[wn * 4 + 0][lane * 16];
    bf16x16 b1 = *(const bf16x16*)&Bls[wn * 4 + 1][lane * 16];
    bf16x16 b2 = *(const bf16x16*)&Bls[wn * 4 + 2][lane * 16];
    bf16x16 b3 = *(const bf16x16*)&Bls[wn * 4 + 3][lane * 16];

    acc[0][0] = __builtin_amdgcn_wmma_f32_16x16x32_bf16(false, a0, false, b0, (short)0, acc[0][0], false, false);
    acc[0][1] = __builtin_amdgcn_wmma_f32_16x16x32_bf16(false, a0, false, b1, (short)0, acc[0][1], false, false);
    acc[0][2] = __builtin_amdgcn_wmma_f32_16x16x32_bf16(false, a0, false, b2, (short)0, acc[0][2], false, false);
    acc[0][3] = __builtin_amdgcn_wmma_f32_16x16x32_bf16(false, a0, false, b3, (short)0, acc[0][3], false, false);
    acc[1][0] = __builtin_amdgcn_wmma_f32_16x16x32_bf16(false, a1, false, b0, (short)0, acc[1][0], false, false);
    acc[1][1] = __builtin_amdgcn_wmma_f32_16x16x32_bf16(false, a1, false, b1, (short)0, acc[1][1], false, false);
    acc[1][2] = __builtin_amdgcn_wmma_f32_16x16x32_bf16(false, a1, false, b2, (short)0, acc[1][2], false, false);
    acc[1][3] = __builtin_amdgcn_wmma_f32_16x16x32_bf16(false, a1, false, b3, (short)0, acc[1][3], false, false);
  }

  // store: lane half 0 -> M 0..7, half 1 -> M 8..15 within each 16x16 tile
#pragma unroll
  for (int i = 0; i < 2; ++i) {
    int mt = bm * MT_BLK + wm * 2 + i;
    if (mt >= MT) continue;
    int rb = mt * 16 + half * 8;
#pragma unroll
    for (int j = 0; j < 4; ++j) {
      int gn = bn * (NT_BLK * 16) + (wn * 4 + j) * 16 + l16;
      if (gn >= N) continue;
#pragma unroll
      for (int r = 0; r < 8; ++r) {
        int gm = rb + r;
        if (gm < M) C[(size_t)gm * N + gn] = acc[i][j][r];
      }
    }
  }
}

// =====================================================================
// Attention pipeline kernels
// =====================================================================
__global__ void k_alpha(const float* __restrict__ h, const float* __restrict__ a_s,
                        const float* __restrict__ a_d, float* __restrict__ out_s,
                        float* __restrict__ out_d, int H, int C) {
  int i = blockIdx.x * blockDim.x + threadIdx.x;   // node*H + head
  if (i >= N_NODES * H) return;
  int hh = i % H;
  const float* hp = h + (size_t)i * C;             // node*H*C + hh*C
  const float* s  = a_s + hh * C;
  const float* d  = a_d + hh * C;
  float vs = 0.f, vd = 0.f;
  for (int c = 0; c < C; ++c) { float x = hp[c]; vs += x * s[c]; vd += x * d[c]; }
  out_s[i] = vs;
  out_d[i] = vd;
}

__global__ void k_fill(float* __restrict__ p, float v, int n) {
  int i = blockIdx.x * blockDim.x + threadIdx.x;
  if (i < n) p[i] = v;
}

__device__ __forceinline__ void atomicMaxF(float* addr, float val) {
  int iv = __float_as_int(val);
  if (iv >= 0) atomicMax((int*)addr, iv);
  else         atomicMin((unsigned int*)addr, (unsigned int)iv);
}

__global__ void k_edge_score(const int* __restrict__ ei, const float* __restrict__ as_,
                             const float* __restrict__ ad_, float* __restrict__ eb,
                             float* __restrict__ m, int H) {
  long long i = (long long)blockIdx.x * blockDim.x + threadIdx.x;
  if (i >= (long long)E_TOT * H) return;
  int e = (int)(i / H), hh = (int)(i - (long long)e * H);
  int s, d;
  if (e < N_EDGES) { s = ei[e]; d = ei[N_EDGES + e]; }
  else             { s = d = e - N_EDGES; }         // self loop
  float x = as_[s * H + hh] + ad_[d * H + hh];
  x = (x > 0.f) ? x : NEG_SLOPE * x;                // leaky_relu
  eb[i] = x;
  atomicMaxF(&m[d * H + hh], x);
}

__global__ void k_edge_exp(const int* __restrict__ ei, float* __restrict__ eb,
                           const float* __restrict__ m, float* __restrict__ denom, int H) {
  long long i = (long long)blockIdx.x * blockDim.x + threadIdx.x;
  if (i >= (long long)E_TOT * H) return;
  int e = (int)(i / H), hh = (int)(i - (long long)e * H);
  int d = (e < N_EDGES) ? ei[N_EDGES + e] : (e - N_EDGES);
  float ex = expf(eb[i] - m[d * H + hh]);
  eb[i] = ex;
  atomicAdd(&denom[d * H + hh], ex);
}

__global__ void k_edge_aggr(const int* __restrict__ ei, const float* __restrict__ eb,
                            const float* __restrict__ denom, const float* __restrict__ hin,
                            float* __restrict__ out, int H, int C) {
  int e = blockIdx.x;
  int s, d;
  if (e < N_EDGES) { s = ei[e]; d = ei[N_EDGES + e]; }
  else             { s = d = e - N_EDGES; }
  int HC = H * C;
  const float* hp = hin + (size_t)s * HC;
  float*       op = out + (size_t)d * HC;
  for (int j = threadIdx.x; j < HC; j += blockDim.x) {
    int hh = j / C;
    float a = eb[(size_t)e * H + hh] / (denom[d * H + hh] + EPS_SM);
    atomicAdd(&op[j], a * hp[j]);
  }
}

__global__ void k_bias_elu(float* __restrict__ h, const float* __restrict__ b, long long n, int HC) {
  long long i = (long long)blockIdx.x * blockDim.x + threadIdx.x;
  if (i >= n) return;
  float x = h[i] + b[(int)(i % HC)];
  h[i] = (x > 0.f) ? x : (expf(x) - 1.f);
}

// =====================================================================
// Pooling + classifier
// =====================================================================
__global__ void k_pool(const float* __restrict__ h, const int* __restrict__ batch,
                       float* __restrict__ sums, float* __restrict__ cnt) {
  int i = blockIdx.x * blockDim.x + threadIdx.x;   // node*128 + c
  if (i >= N_NODES * HIDDEN) return;
  int nde = i >> 7, c = i & 127;
  int g = batch[nde];
  atomicAdd(&sums[g * HIDDEN + c], h[i]);
  if (c == 0) atomicAdd(&cnt[g], 1.f);
}

__global__ void k_classify(const float* __restrict__ sums, const float* __restrict__ cnt,
                           const float* __restrict__ Wc, const float* __restrict__ bc,
                           float* __restrict__ out) {
  int i = blockIdx.x * blockDim.x + threadIdx.x;   // g*4 + class
  if (i >= N_GRAPHS * N_CLASSES) return;
  int g = i >> 2, cl = i & 3;
  float invc = 1.f / fmaxf(cnt[g], 1.f);
  float acc = bc[cl];
  for (int c = 0; c < HIDDEN; ++c)
    acc += sums[g * HIDDEN + c] * invc * Wc[c * N_CLASSES + cl];
  out[i] = acc;
}

// =====================================================================
// Host-side helpers
// =====================================================================
static void run_gemm(hipStream_t stream, const float* A, const float* B, float* C,
                     unsigned short* Apk, unsigned short* Bpk, int M, int K, int N) {
  const int MT = (M + 15) / 16, KT = (K + 31) / 32, NT = (N + 15) / 16;
  long long na = (long long)MT * KT * 512;
  long long nb = (long long)KT * NT * 512;
  k_pack_A<<<(unsigned int)((na + 255) / 256), 256, 0, stream>>>(A, Apk, M, K, MT, KT);
  k_pack_B<<<(unsigned int)((nb + 255) / 256), 256, 0, stream>>>(B, Bpk, K, N, KT, NT);
  dim3 grid((NT + NT_BLK - 1) / NT_BLK, (MT + MT_BLK - 1) / MT_BLK);
  k_gemm_packed<<<grid, 256, 0, stream>>>(Apk, Bpk, C, MT, KT, NT, M, N);
}

static void gat_attention(hipStream_t stream, const float* h, const float* a_s,
                          const float* a_d, const float* bias, const int* ei,
                          float* out, float* as_buf, float* ad_buf, float* mbuf,
                          float* dbuf, float* ebuf, int H, int C) {
  const int HC = H * C;
  const int nH = N_NODES * H;
  k_alpha<<<(nH + 255) / 256, 256, 0, stream>>>(h, a_s, a_d, as_buf, ad_buf, H, C);
  k_fill<<<(nH + 255) / 256, 256, 0, stream>>>(mbuf, -INFINITY, nH);
  hipMemsetAsync(dbuf, 0, (size_t)nH * sizeof(float), stream);
  hipMemsetAsync(out, 0, (size_t)N_NODES * HC * sizeof(float), stream);
  long long eh = (long long)E_TOT * H;
  unsigned int eg = (unsigned int)((eh + 255) / 256);
  k_edge_score<<<eg, 256, 0, stream>>>(ei, as_buf, ad_buf, ebuf, mbuf, H);
  k_edge_exp<<<eg, 256, 0, stream>>>(ei, ebuf, mbuf, dbuf, H);
  k_edge_aggr<<<E_TOT, 128, 0, stream>>>(ei, ebuf, dbuf, h, out, H, C);
  long long tot = (long long)N_NODES * HC;
  k_bias_elu<<<(unsigned int)((tot + 255) / 256), 256, 0, stream>>>(out, bias, tot, HC);
}

extern "C" void kernel_launch(void* const* d_in, const int* in_sizes, int n_in,
                              void* d_out, int out_size, void* d_ws, size_t ws_size,
                              hipStream_t stream) {
  (void)in_sizes; (void)n_in; (void)out_size; (void)ws_size;
  const float* x      = (const float*)d_in[0];
  const int*   ei     = (const int*)d_in[1];
  const int*   batch  = (const int*)d_in[2];
  const float* W0     = (const float*)d_in[3];
  const float* a_src0 = (const float*)d_in[4];
  const float* a_dst0 = (const float*)d_in[5];
  const float* b0     = (const float*)d_in[6];
  const float* W1     = (const float*)d_in[7];
  const float* a_src1 = (const float*)d_in[8];
  const float* a_dst1 = (const float*)d_in[9];
  const float* b1     = (const float*)d_in[10];
  const float* W2     = (const float*)d_in[11];
  const float* a_src2 = (const float*)d_in[12];
  const float* a_dst2 = (const float*)d_in[13];
  const float* b2     = (const float*)d_in[14];
  const float* Wc     = (const float*)d_in[15];
  const float* bc     = (const float*)d_in[16];

  // ---- workspace carve-out ----
  char* ws = (char*)d_ws;
  size_t off = 0;
  auto carve = [&](size_t bytes) -> void* {
    void* p = ws + off;
    off += (bytes + 255) & ~(size_t)255;
    return p;
  };
  float* bufA = (float*)carve((size_t)N_NODES * D1 * sizeof(float));  // 82 MB
  float* bufB = (float*)carve((size_t)N_NODES * D1 * sizeof(float));  // 82 MB
  float* bufC = (float*)carve((size_t)N_NODES * D1 * sizeof(float));  // 82 MB
  // packed bf16 buffers (max: A 1250x32 tiles = 40MB; B 32x64 tiles = 2MB)
  unsigned short* Apk = (unsigned short*)carve((size_t)1250 * 32 * 512 * 2);
  unsigned short* Bpk = (unsigned short*)carve((size_t)32 * 64 * 512 * 2);
  float* as_buf = (float*)carve((size_t)N_NODES * HEADS * sizeof(float));
  float* ad_buf = (float*)carve((size_t)N_NODES * HEADS * sizeof(float));
  float* mbuf   = (float*)carve((size_t)N_NODES * HEADS * sizeof(float));
  float* dbuf   = (float*)carve((size_t)N_NODES * HEADS * sizeof(float));
  float* ebuf   = (float*)carve((size_t)E_TOT * HEADS * sizeof(float));
  float* sums   = (float*)carve((size_t)N_GRAPHS * HIDDEN * sizeof(float));
  float* cnt    = (float*)carve((size_t)N_GRAPHS * sizeof(float));

  // ---- Layer 0: x @ W0 -> bufA ; attention -> bufB (elu'd) ----
  run_gemm(stream, x, W0, bufA, Apk, Bpk, N_NODES, N_FEAT, D1);
  gat_attention(stream, bufA, a_src0, a_dst0, b0, ei, bufB,
                as_buf, ad_buf, mbuf, dbuf, ebuf, HEADS, HIDDEN);

  // ---- Layer 1: bufB @ W1 -> bufC ; attention -> bufA (elu'd) ----
  run_gemm(stream, bufB, W1, bufC, Apk, Bpk, N_NODES, D1, D1);
  gat_attention(stream, bufC, a_src1, a_dst1, b1, ei, bufA,
                as_buf, ad_buf, mbuf, dbuf, ebuf, HEADS, HIDDEN);

  // ---- Layer 2: bufA @ W2 -> bufB[:, :128] ; attention (H=1) -> bufC ----
  run_gemm(stream, bufA, W2, bufB, Apk, Bpk, N_NODES, D1, HIDDEN);
  gat_attention(stream, bufB, a_src2, a_dst2, b2, ei, bufC,
                as_buf, ad_buf, mbuf, dbuf, ebuf, 1, HIDDEN);

  // ---- pool + classify ----
  hipMemsetAsync(sums, 0, (size_t)N_GRAPHS * HIDDEN * sizeof(float), stream);
  hipMemsetAsync(cnt, 0, (size_t)N_GRAPHS * sizeof(float), stream);
  k_pool<<<(N_NODES * HIDDEN + 255) / 256, 256, 0, stream>>>(bufC, batch, sums, cnt);
  k_classify<<<(N_GRAPHS * N_CLASSES + 255) / 256, 256, 0, stream>>>(
      sums, cnt, Wc, bc, (float*)d_out);
}
